// GRAND_89859305766914
// MI455X (gfx1250) — matrix operand order repaired and checked by reading, not compiled, not run on hardware
//
#include <hip/hip_runtime.h>

#define N_NODES 100000
#define N_EDGES 1600000
#define D_IN    128
#define D_HID   128
#define D_OUT   64
#define K_PROP  4

typedef __bf16 bf16;
typedef __attribute__((ext_vector_type(16))) __bf16 v16bf;
typedef __attribute__((ext_vector_type(8)))  float  v8f;

// ---------------- degree / normalization ----------------

__global__ void k_deg(const int* __restrict__ ei, float* __restrict__ deg) {
    int e = blockIdx.x * blockDim.x + threadIdx.x;
    if (e >= N_EDGES) return;
    int d = ei[N_EDGES + e];
    unsafeAtomicAdd(&deg[d], 1.0f);
}

__global__ void k_norm(float* deg) {
    int i = blockIdx.x * blockDim.x + threadIdx.x;
    if (i >= N_NODES) return;
    float d = deg[i];
    deg[i] = rsqrtf(d > 1.0f ? d : 1.0f);
}

// ---------------- init: x = y = feats * (1 - DROPOUT), float4-wide ----------------

__global__ void k_init(const float4* __restrict__ feats,
                       float4* __restrict__ x, float4* __restrict__ y) {
    size_t i = (size_t)blockIdx.x * blockDim.x + threadIdx.x;
    if (i >= (size_t)N_NODES * D_IN / 4) return;
    float4 f = feats[i];
    float4 v = make_float4(f.x * 0.5f, f.y * 0.5f, f.z * 0.5f, f.w * 0.5f);
    x[i] = v;
    y[i] = v;
}

// ---------------- one propagation round: xn[dst] += w * x[src] ----------------
// One wave (32 lanes) per edge; each lane moves a float4 (128 floats total).
// Edge index is wave-uniform: readfirstlane scalarizes the src/dst/norm loads
// (s_load via constant cache) and enables SGPR-base global addressing.

__global__ void k_scatter(const int* __restrict__ ei, const float* __restrict__ norm,
                          const float* __restrict__ x, float* __restrict__ xn) {
    unsigned gid  = blockIdx.x * blockDim.x + threadIdx.x;
    unsigned e    = (unsigned)__builtin_amdgcn_readfirstlane((int)(gid >> 5));
    unsigned lane = threadIdx.x & 31u;
    if (e >= N_EDGES) return;
    int s = ei[e];                 // uniform -> scalar load
    int d = ei[N_EDGES + e];       // uniform -> scalar load
    float w = norm[s] * norm[d];   // uniform -> scalar loads
    const float4 v = *(const float4*)(x + (size_t)s * D_IN + lane * 4);
    float* o = xn + (size_t)d * D_IN + lane * 4;
    unsafeAtomicAdd(o + 0, v.x * w);
    unsafeAtomicAdd(o + 1, v.y * w);
    unsafeAtomicAdd(o + 2, v.z * w);
    unsafeAtomicAdd(o + 3, v.w * w);
}

__global__ void k_accum(const float4* __restrict__ xn, float4* __restrict__ y) {
    size_t i = (size_t)blockIdx.x * blockDim.x + threadIdx.x;
    if (i >= (size_t)N_NODES * D_IN / 4) return;
    float4 a = xn[i];
    float4 b = y[i];
    y[i] = make_float4(a.x + b.x, a.y + b.y, a.z + b.z, a.w + b.w);
}

// ---------------- weight convert + transpose: W[k][n] -> Wt[n][k] bf16 ----------------

__global__ void k_cvtW(const float* __restrict__ W, bf16* __restrict__ Wt,
                       int kdim, int ndim) {
    int i = blockIdx.x * blockDim.x + threadIdx.x;
    if (i >= kdim * ndim) return;
    int k = i / ndim, n = i % ndim;
    Wt[(size_t)n * kdim + k] = (bf16)W[i];
}

// ---------------- fused MLP: out = relu((y/5) @ W1) @ W2 ----------------
// 4 waves per block; each wave owns a 16-row node block. Weights are staged
// once per workgroup in LDS (shared by all waves); B fragments come from
// ds_load. Layer-1 output is staged per-wave in LDS (bf16, row-major) to
// convert the C/D fragment layout into the A fragment layout for layer 2.

__global__ void __launch_bounds__(128) k_gemm(const float* __restrict__ Y,
                                              const bf16* __restrict__ W1t,
                                              const bf16* __restrict__ W2t,
                                              float* __restrict__ Out) {
    __shared__ bf16 w1s[D_HID * D_IN];    // 32 KB, [n][k]
    __shared__ bf16 w2s[D_OUT * D_HID];   // 16 KB, [n][k]
    __shared__ bf16 hstage[4][16 * D_HID];// 16 KB

    const int tid = threadIdx.x;
    // cooperative weight staging (uint4 = 16B chunks)
    {
        const uint4* s1 = (const uint4*)W1t;
        uint4*       d1 = (uint4*)w1s;
        #pragma unroll
        for (int i = 0; i < (D_HID * D_IN * 2 / 16) / 128; ++i)   // 2048/128 = 16
            d1[tid + i * 128] = s1[tid + i * 128];
        const uint4* s2 = (const uint4*)W2t;
        uint4*       d2 = (uint4*)w2s;
        #pragma unroll
        for (int i = 0; i < (D_OUT * D_HID * 2 / 16) / 128; ++i)  // 1024/128 = 8
            d2[tid + i * 128] = s2[tid + i * 128];
    }
    __syncthreads();   // all 128 threads reach this before any divergent exit

    const int wave = tid >> 5;
    const int lane = tid & 31;
    const int mblk = blockIdx.x * 4 + wave;
    if (mblk >= N_NODES / 16) return;
    const int row  = lane & 15;   // M for A / N for B,C
    const int half = lane >> 4;

    // --- A fragments: 16x128 block of y, scaled by 1/(K+1), cast to bf16 ---
    // A 16x32 order used by this code: lanes 0-15 hold M=lane; elems 0-15 -> K=half*16+e
    // (contiguous, matching B).
    v16bf a[4];
    const float inv = 1.0f / (float)(K_PROP + 1);
    const float* yrow = Y + ((size_t)mblk * 16 + row) * D_IN;
    #pragma unroll
    for (int kt = 0; kt < 4; ++kt) {
        const float* p = yrow + kt * 32 + half * 16;
        #pragma unroll
        for (int e = 0; e < 8; ++e) {
            a[kt][e]     = (bf16)(p[e] * inv);
            a[kt][e + 8] = (bf16)(p[e + 8] * inv);
        }
    }

    bf16* hs = &hstage[wave][0];

    // --- layer 1: H = relu(A @ W1), 8 N-tiles x 4 K-steps, B from LDS ---
    #pragma unroll
    for (int nt = 0; nt < 8; ++nt) {
        v8f c = {};
        #pragma unroll
        for (int kt = 0; kt < 4; ++kt) {
            // B 32x16 as loaded by this code: lane = col N, half selects K block of 16 -> contiguous
            v16bf b = *(const v16bf*)(w1s + (nt * 16 + row) * D_HID + kt * 32 + half * 16);
            c = __builtin_amdgcn_wmma_f32_16x16x32_bf16(false, a[kt], false, b,
                                                        (short)0, c, false, false);
        }
        // C/D layout: VGPR g -> M = half*8+g, N = lane&15
        #pragma unroll
        for (int g = 0; g < 8; ++g) {
            float v = c[g];
            v = v > 0.0f ? v : 0.0f;
            hs[(half * 8 + g) * D_HID + nt * 16 + row] = (bf16)v;
        }
    }
    // hstage slab is wave-private; same-wave DS ops complete in order.

    // --- reload H as A fragments for layer 2 ---
    v16bf a2[4];
    #pragma unroll
    for (int kt = 0; kt < 4; ++kt) {
        const bf16* p = hs + row * D_HID + kt * 32 + half * 16;
        #pragma unroll
        for (int e = 0; e < 8; ++e) {
            a2[kt][e]     = p[e];
            a2[kt][e + 8] = p[e + 8];
        }
    }

    // --- layer 2: Out = H @ W2, 4 N-tiles x 4 K-steps, B from LDS ---
    #pragma unroll
    for (int nt = 0; nt < 4; ++nt) {
        v8f c = {};
        #pragma unroll
        for (int kt = 0; kt < 4; ++kt) {
            v16bf b = *(const v16bf*)(w2s + (nt * 16 + row) * D_HID + kt * 32 + half * 16);
            c = __builtin_amdgcn_wmma_f32_16x16x32_bf16(false, a2[kt], false, b,
                                                        (short)0, c, false, false);
        }
        #pragma unroll
        for (int g = 0; g < 8; ++g) {
            Out[((size_t)mblk * 16 + half * 8 + g) * D_OUT + nt * 16 + row] = c[g];
        }
    }
}

// ---------------- host-side orchestration ----------------

extern "C" void kernel_launch(void* const* d_in, const int* in_sizes, int n_in,
                              void* d_out, int out_size, void* d_ws, size_t ws_size,
                              hipStream_t stream) {
    const float* feats = (const float*)d_in[0];
    const float* W1    = (const float*)d_in[1];
    const float* W2    = (const float*)d_in[2];
    const int*   ei    = (const int*)d_in[3];
    float* out = (float*)d_out;

    const size_t NF_BYTES = (size_t)N_NODES * D_IN * sizeof(float); // 51.2 MB
    char* ws = (char*)d_ws;
    size_t off = 0;
    auto carve = [&](size_t bytes) { char* p = ws + off; off += (bytes + 255) & ~(size_t)255; return p; };

    float* norm = (float*)carve((size_t)N_NODES * sizeof(float));
    float* bufA = (float*)carve(NF_BYTES);
    float* bufB = (float*)carve(NF_BYTES);
    float* bufY = (float*)carve(NF_BYTES);
    bf16*  W1t  = (bf16*)carve((size_t)D_IN * D_HID * sizeof(bf16));
    bf16*  W2t  = (bf16*)carve((size_t)D_HID * D_OUT * sizeof(bf16));

    // degree -> norm
    hipMemsetAsync(norm, 0, (size_t)N_NODES * sizeof(float), stream);
    k_deg<<<(N_EDGES + 255) / 256, 256, 0, stream>>>(ei, norm);
    k_norm<<<(N_NODES + 255) / 256, 256, 0, stream>>>(norm);

    // x = y = feats * 0.5   (float4-wide)
    const int NV4 = (int)((size_t)N_NODES * D_IN / 4);
    const int NV4_BLOCKS = (NV4 + 255) / 256;
    k_init<<<NV4_BLOCKS, 256, 0, stream>>>((const float4*)feats, (float4*)bufA, (float4*)bufY);

    // weight conversion (independent of propagation; runs early on the stream)
    k_cvtW<<<(D_IN * D_HID + 255) / 256, 256, 0, stream>>>(W1, W1t, D_IN, D_HID);
    k_cvtW<<<(D_HID * D_OUT + 255) / 256, 256, 0, stream>>>(W2, W2t, D_HID, D_OUT);

    // K rounds of normalized propagation with running sum
    float* x  = bufA;
    float* xn = bufB;
    const int SCATTER_BLOCKS = (int)(((size_t)N_EDGES * 32 + 255) / 256);
    for (int k = 0; k < K_PROP; ++k) {
        hipMemsetAsync(xn, 0, NF_BYTES, stream);
        k_scatter<<<SCATTER_BLOCKS, 256, 0, stream>>>(ei, norm, x, xn);
        k_accum<<<NV4_BLOCKS, 256, 0, stream>>>((const float4*)xn, (float4*)bufY);
        float* t = x; x = xn; xn = t;
    }

    // fused 2-layer MLP with WMMA
    const int MBLKS = N_NODES / 16;              // 6250
    k_gemm<<<(MBLKS + 3) / 4, 128, 0, stream>>>(bufY, W1t, W2t, out);
}
